// BasicGcn_11295763988680
// MI455X (gfx1250) — matrix operand order repaired and checked by reading, not compiled, not run on hardware
//
#include <hip/hip_runtime.h>
#include <hip/hip_bf16.h>

typedef __attribute__((ext_vector_type(16))) _Float16 v16h;
typedef __attribute__((ext_vector_type(8)))  _Float16 v8h;
typedef __attribute__((ext_vector_type(8)))  float    v8f;

// ---------------------------------------------------------------------------
// Degree / normalization kernels
// ---------------------------------------------------------------------------
__global__ void fill_f32_kernel(float* __restrict__ p, long long n, float v) {
    long long i = (long long)blockIdx.x * blockDim.x + threadIdx.x;
    long long stride = (long long)gridDim.x * blockDim.x;
    for (; i < n; i += stride) p[i] = v;
}

__global__ void accum_deg_kernel(const long long* __restrict__ ei,
                                 const float* __restrict__ ew,
                                 float* __restrict__ deg, long long E) {
    long long i = (long long)blockIdx.x * blockDim.x + threadIdx.x;
    long long stride = (long long)gridDim.x * blockDim.x;
    for (; i < E; i += stride) {
        long long d = ei[E + i];          // dst row of edge_index
        atomicAdd(&deg[d], ew[i]);
    }
}

__global__ void dinv_kernel(float* __restrict__ deg, long long n) {
    long long i = (long long)blockIdx.x * blockDim.x + threadIdx.x;
    long long stride = (long long)gridDim.x * blockDim.x;
    for (; i < n; i += stride) {
        float d = deg[i];
        deg[i] = (d > 0.0f) ? rsqrtf(d) : 0.0f;   // in-place deg -> dinv
    }
}

// ---------------------------------------------------------------------------
// WMMA GEMM:  C[N, OUTD] = A[N, 128] (f32 -> f16) @ W[128, OUTD] (f32 -> f16)
// Block = 256 threads (8 waves) = one 16-row M-tile. The A-tile is staged
// once per block into LDS as f16 (4.3 KB incl. padding), then each wave
// computes one 16x16 column tile with 4 x v_wmma_f32_16x16x32_f16 over K=128.
// A-fragment reads from LDS are two contiguous 16B chunks -> ds_load_b128.
// Fragment layouts per CDNA5 ISA 7.12.2 (wave32).
// ---------------------------------------------------------------------------
template <int OUTD>
__global__ void gemm_wmma_kernel(const float* __restrict__ A,
                                 const float* __restrict__ W,
                                 float* __restrict__ C, int nrows) {
    constexpr int TN  = OUTD / 16;               // 16-col tiles (8 or 4)
    constexpr int LDA = 128 + 8;                 // padded halfs per LDS row
    __shared__ _Float16 As[16 * LDA];

    const int tm   = blockIdx.x;
    const int lane = threadIdx.x & 31;
    const int wave = threadIdx.x >> 5;

    // ---- cooperative stage: 16 rows x 128 f32 -> f16 in LDS (once) ----
    {
        const int r = threadIdx.x >> 4;          // 0..15
        const int c = (threadIdx.x & 15) * 8;    // 0,8,...,120
        int gr = tm * 16 + r;
        if (gr >= nrows) gr = nrows - 1;         // clamp (stores are guarded)
        const float4 f0 = *(const float4*)(A + (long long)gr * 128 + c);
        const float4 f1 = *(const float4*)(A + (long long)gr * 128 + c + 4);
        _Float16* dst = &As[r * LDA + c];        // 16B-aligned (272*r + 2c)
        dst[0] = (_Float16)f0.x; dst[1] = (_Float16)f0.y;
        dst[2] = (_Float16)f0.z; dst[3] = (_Float16)f0.w;
        dst[4] = (_Float16)f1.x; dst[5] = (_Float16)f1.y;
        dst[6] = (_Float16)f1.z; dst[7] = (_Float16)f1.w;
    }
    __syncthreads();

    if (wave >= TN) return;                      // wave-uniform exit (OUTD=64)
    const int tn = wave;

    // A fragment: row = lane&15; lanes 0-15 take K-halves 0..7, lanes 16-31
    // take K-halves 8..15 of each 16-K group -> two contiguous 8-half runs.
    const int arow = lane & 15;
    const int hk   = (lane >> 4) * 8;

    // B fragment: col = lane&15; lanes 0-15 hold K=0..7,16..23, lanes 16-31
    // hold K=8..15,24..31 of each 32-K step (matches A).
    const int col  = tn * 16 + (lane & 15);
    const int kofB = (lane >> 4) * 8;

    v8f acc = {};
#pragma unroll
    for (int ks = 0; ks < 4; ++ks) {
        const int k0 = ks * 32;
        v16h a, b;
        const v8h a0 = *(const v8h*)&As[arow * LDA + k0 + hk];        // K k0+hk..+7
        const v8h a1 = *(const v8h*)&As[arow * LDA + k0 + 16 + hk];   // K k0+16+hk..+7
#pragma unroll
        for (int j = 0; j < 8; ++j) { a[j] = a0[j]; a[8 + j] = a1[j]; }
#pragma unroll
        for (int j = 0; j < 16; ++j)
            b[j] = (_Float16)W[(k0 + kofB + j + (j >= 8 ? 8 : 0)) * OUTD + col];
        // (neg_a, A, neg_b, B, c_mod, C, reuse_a, reuse_b)
        acc = __builtin_amdgcn_wmma_f32_16x16x32_f16(
            false, a, false, b, (short)0, acc, false, false);
    }

    // D layout: col = lane&15, row = j + 8*(lane>>4)
    const int mbase = tm * 16 + (lane >> 4) * 8;
#pragma unroll
    for (int j = 0; j < 8; ++j) {
        const int r2 = mbase + j;
        if (r2 < nrows) C[(long long)r2 * OUTD + col] = acc[j];
    }
}

// ---------------------------------------------------------------------------
// out[n, c] = bias[c] + dinv[n]^2 * xw[n, c]     (self-loop + bias folded)
// ---------------------------------------------------------------------------
template <int OUTD>
__global__ void init_out_kernel(const float* __restrict__ xw,
                                const float* __restrict__ dinv,
                                const float* __restrict__ bias,
                                float* __restrict__ out, int nrows) {
    long long total  = (long long)nrows * OUTD;
    long long i      = (long long)blockIdx.x * blockDim.x + threadIdx.x;
    long long stride = (long long)gridDim.x * blockDim.x;
    for (; i < total; i += stride) {
        const int n = (int)(i / OUTD);
        const int c = (int)(i % OUTD);
        const float dv = dinv[n];
        out[i] = bias[c] + dv * dv * xw[i];
    }
}

// ---------------------------------------------------------------------------
// Edge scatter: one wave32 per edge.  Lane gathers float4 (OUTD=128) or
// float2 (OUTD=64) of xw[src], scales by norm, atomic-adds into out[dst].
// xw (51 MB) is L2-resident on MI455X (192 MB L2) -> gathers are L2 hits.
// ---------------------------------------------------------------------------
template <int OUTD>
__global__ void scatter_kernel(const long long* __restrict__ ei,
                               const float* __restrict__ ew,
                               const float* __restrict__ dinv,
                               const float* __restrict__ xw,
                               float* __restrict__ out, long long E) {
    constexpr int VEC = OUTD / 32;                // floats per lane (4 or 2)
    const int lane = threadIdx.x & 31;
    const int wave = threadIdx.x >> 5;
    const int wpb  = blockDim.x >> 5;
    long long e      = (long long)blockIdx.x * wpb + wave;
    const long long stride = (long long)gridDim.x * wpb;

    for (; e < E; e += stride) {
        const long long s = ei[e];
        const long long d = ei[E + e];
        const float norm  = dinv[s] * ew[e] * dinv[d];

        const float* __restrict__ xs = xw + s * OUTD + lane * VEC;
        float* __restrict__ od       = out + d * OUTD + lane * VEC;

        if (VEC == 4) {
            const float4 v = *(const float4*)xs;
            atomicAdd(od + 0, norm * v.x);
            atomicAdd(od + 1, norm * v.y);
            atomicAdd(od + 2, norm * v.z);
            atomicAdd(od + 3, norm * v.w);
        } else {
            const float2 v = *(const float2*)xs;
            atomicAdd(od + 0, norm * v.x);
            atomicAdd(od + 1, norm * v.y);
        }
    }
}

__global__ void relu_kernel(float* __restrict__ p, long long n) {
    long long i = (long long)blockIdx.x * blockDim.x + threadIdx.x;
    long long stride = (long long)gridDim.x * blockDim.x;
    for (; i < n; i += stride) {
        const float v = p[i];
        p[i] = v > 0.0f ? v : 0.0f;
    }
}

// ---------------------------------------------------------------------------
// Host-side orchestration
// ---------------------------------------------------------------------------
extern "C" void kernel_launch(void* const* d_in, const int* in_sizes, int n_in,
                              void* d_out, int out_size, void* d_ws, size_t ws_size,
                              hipStream_t stream) {
    const float*     x  = (const float*)d_in[0];
    const long long* ei = (const long long*)d_in[1];   // int64 edge_index [2, E]
    const float*     ew = (const float*)d_in[2];
    const float*     W1 = (const float*)d_in[3];
    const float*     b1 = (const float*)d_in[4];
    const float*     W2 = (const float*)d_in[5];
    const float*     b2 = (const float*)d_in[6];
    const float*     W3 = (const float*)d_in[7];
    const float*     b3 = (const float*)d_in[8];

    const int       N = in_sizes[0] / 128;    // 100000 nodes
    const long long E = in_sizes[1] / 2;      // 3.2M edges

    // Workspace layout (256B aligned): bufA (XW), bufB (H), dinv
    char* ws = (char*)d_ws;
    const size_t matBytes = ((size_t)N * 128 * sizeof(float) + 255) & ~(size_t)255;
    float* bufA = (float*)(ws);
    float* bufB = (float*)(ws + matBytes);
    float* dinv = (float*)(ws + 2 * matBytes);

    float* out = (float*)d_out;               // [N, 64] f32

    const int EB = 256;                       // elementwise block
    const int eBlocks = 4096;

    // ---- degree -> dinv (self-loop weight 1.0 folded into the init) ----
    fill_f32_kernel<<<1024, EB, 0, stream>>>(dinv, N, 1.0f);
    accum_deg_kernel<<<eBlocks, EB, 0, stream>>>(ei, ew, dinv, E);
    dinv_kernel<<<1024, EB, 0, stream>>>(dinv, N);

    // ---- launch geometry ----
    const int tiles_m = (N + 15) >> 4;        // 6250 M-tiles -> one block each
    const long long nh = (long long)N * 128;
    const int sBlocks = 8192;                 // 65536 waves for 3.2M edges

    // ---- Layer 1: x @ W1 -> bufA; aggregate -> bufB; relu ----
    gemm_wmma_kernel<128><<<tiles_m, 256, 0, stream>>>(x, W1, bufA, N);
    init_out_kernel<128><<<eBlocks, EB, 0, stream>>>(bufA, dinv, b1, bufB, N);
    scatter_kernel<128><<<sBlocks, 256, 0, stream>>>(ei, ew, dinv, bufA, bufB, E);
    relu_kernel<<<eBlocks, EB, 0, stream>>>(bufB, nh);

    // ---- Layer 2: bufB @ W2 -> bufA; aggregate -> bufB; relu ----
    gemm_wmma_kernel<128><<<tiles_m, 256, 0, stream>>>(bufB, W2, bufA, N);
    init_out_kernel<128><<<eBlocks, EB, 0, stream>>>(bufA, dinv, b2, bufB, N);
    scatter_kernel<128><<<sBlocks, 256, 0, stream>>>(ei, ew, dinv, bufA, bufB, E);
    relu_kernel<<<eBlocks, EB, 0, stream>>>(bufB, nh);

    // ---- Layer 3: bufB @ W3 -> bufA (out=64); aggregate -> d_out ----
    gemm_wmma_kernel<64><<<tiles_m, 256, 0, stream>>>(bufB, W3, bufA, N);
    init_out_kernel<64><<<eBlocks, EB, 0, stream>>>(bufA, dinv, b3, out, N);
    scatter_kernel<64><<<sBlocks, 256, 0, stream>>>(ei, ew, dinv, bufA, out, E);
}
